// CachedSelfAttention_37572373905679
// MI455X (gfx1250) — matrix operand order repaired and checked by reading, not compiled, not run on hardware
//
#include <hip/hip_runtime.h>
#include <hip/hip_bf16.h>
#include <math.h>

// ---------------------------------------------------------------------------
// CachedSelfAttention for MI455X (gfx1250, wave32, WMMA bf16)
// B=2, S=2048, P=2048, D=2048, H=16, HD=128.  Compute-bound (~275 GFLOP vs
// ~9us of HBM traffic) -> convert everything to bf16 once and run all four
// GEMMs + attention through v_wmma_f32_16x16x32_bf16 with f32 accumulation.
// GEMM uses 64x64 per-wave tiles: 16 b128 loads per 16 WMMAs (1.0 load/WMMA).
// ---------------------------------------------------------------------------

typedef __attribute__((ext_vector_type(16))) __bf16 v16bf;
typedef __attribute__((ext_vector_type(8)))  float  v8f;

#define DEVINL __device__ __forceinline__

constexpr int Bc  = 2;
constexpr int Sc  = 2048;
constexpr int Pc  = 2048;
constexpr int Dc  = 2048;
constexpr int Hc  = 16;
constexpr int HDc = 128;
constexpr int KPc = Pc + Sc;   // 4096 total keys
constexpr int Mc  = Bc * Sc;   // 4096 rows in projection GEMMs

union BF16Frag { v16bf v; unsigned u[8]; };

DEVINL unsigned short f32_to_bf16(float f) {
  unsigned u = __float_as_uint(f);
  u += 0x7FFFu + ((u >> 16) & 1u);   // round-to-nearest-even
  return (unsigned short)(u >> 16);
}

DEVINL v8f zero_v8f() {
  v8f z;
#pragma unroll
  for (int i = 0; i < 8; ++i) z[i] = 0.0f;
  return z;
}

#define WMMA_BF16(a, b, c) \
  __builtin_amdgcn_wmma_f32_16x16x32_bf16(false, (a), false, (b), (short)0, (c), false, false)

// A-matrix fragment (16x32 bf16, MxK), source row-major [rows, ld].
// ISA 7.12.2: lanes 0-15 & 16-31 both hold M=0..15; K pairs:
//   half 0: VGPR p<4 -> K=2p,2p+1 ; p>=4 -> K=2p+8 ; half 1: +8.
DEVINL v16bf load_a_frag(const unsigned short* base, int ld, int row0, int k0) {
  const int lane = threadIdx.x & 31;
  const int half = lane >> 4;
  const unsigned short* rp = base + (long)(row0 + (lane & 15)) * ld + k0;
  BF16Frag f;
#pragma unroll
  for (int p = 0; p < 8; ++p) {
    const int k = ((p < 4) ? (2 * p) : (2 * p + 8)) + half * 8;
    f.u[p] = *(const unsigned int*)(rp + k);
  }
  return f.v;
}

// B-matrix fragment (32x16 bf16, KxN): lane holds column N=lane%16; the
// source here is the TRANSPOSED operand stored row-major [N, ld] so each
// lane reads K pairs in the same order as the A fragment (table order).
DEVINL v16bf load_b_frag(const unsigned short* base, int ld, int col0, int k0) {
  const int lane = threadIdx.x & 31;
  const int half = lane >> 4;
  const unsigned short* rp = base + (long)(col0 + (lane & 15)) * ld + k0;
  BF16Frag f;
#pragma unroll
  for (int p = 0; p < 8; ++p) f.u[p] = *(const unsigned int*)(rp + ((p < 4) ? (2 * p) : (2 * p + 8)) + half * 8);
  return f.v;
}

// ---------------------------------------------------------------------------
// Conversion kernels (one-time bf16 staging; bandwidth cost is negligible)
// ---------------------------------------------------------------------------
__global__ void f32_to_bf16_kernel(const float* __restrict__ in,
                                   unsigned short* __restrict__ out, long n) {
  long i = (long)blockIdx.x * blockDim.x + threadIdx.x;
  if (i < n) out[i] = f32_to_bf16(in[i]);
}

// Wt[n, k] = W[k, n]  (bf16) -> B-fragment loads become contiguous
__global__ void transpose_w_kernel(const float* __restrict__ w,
                                   unsigned short* __restrict__ wt) {
  long i = (long)blockIdx.x * blockDim.x + threadIdx.x;
  if (i >= (long)Dc * Dc) return;
  int nn = (int)(i / Dc), kk = (int)(i % Dc);
  wt[i] = f32_to_bf16(w[(long)kk * Dc + nn]);
}

// cache_k[b,p,d] -> Kbf[b, p, d] inside the [B, P+S, D] buffer
__global__ void cache_k_kernel(const float* __restrict__ ck,
                               unsigned short* __restrict__ kb) {
  long i = (long)blockIdx.x * blockDim.x + threadIdx.x;
  if (i >= (long)Bc * Pc * Dc) return;
  int d = (int)(i % Dc);
  long t = i / Dc;
  int p = (int)(t % Pc);
  int b = (int)(t / Pc);
  kb[((long)b * KPc + p) * Dc + d] = f32_to_bf16(ck[i]);
}

// cache_v[b,p,d] -> Vt[b, d, p]  (transposed so P*V B-fragments are contiguous)
__global__ void cache_v_kernel(const float* __restrict__ cv,
                               unsigned short* __restrict__ vt) {
  long i = (long)blockIdx.x * blockDim.x + threadIdx.x;
  if (i >= (long)Bc * Pc * Dc) return;
  int d = (int)(i % Dc);
  long t = i / Dc;
  int p = (int)(t % Pc);
  int b = (int)(t / Pc);
  vt[((long)b * Dc + d) * KPc + p] = f32_to_bf16(cv[i]);
}

// ---------------------------------------------------------------------------
// WMMA GEMM:  out = A[M,K] @ W[K,N] + bias   (A bf16 row-major, W given as
// Wt bf16 [N,K]).  8 waves (2x4), 128x256 block tile, 64x64 per wave:
// per 32-K step: 4 A-frags + 4 B-frags (16 b128 loads) feed 16 WMMAs.
// mode 0: bf16 row-major out (per-batch row offset, for Q / K_new)
// mode 1: bf16 transposed out (for V_new -> Vt[b, n, P+s])
// mode 2: f32 row-major out (final projection into d_out)
// ---------------------------------------------------------------------------
__global__ void __launch_bounds__(256) gemm_bf16(
    const unsigned short* __restrict__ A, const unsigned short* __restrict__ Wt,
    const float* __restrict__ bias, void* __restrict__ out,
    float scale, int mode, long obStride, long oRowOff, long ldOut) {
  const int lane = threadIdx.x & 31;
  const int wid  = threadIdx.x >> 5;        // 0..7
  const int wy = wid >> 2, wx = wid & 3;    // 2 x 4 wave grid
  const int m0 = blockIdx.y * 128 + wy * 64;
  const int n0 = blockIdx.x * 256 + wx * 64;

  v8f acc[4][4];
#pragma unroll
  for (int i = 0; i < 4; ++i)
#pragma unroll
    for (int j = 0; j < 4; ++j) acc[i][j] = zero_v8f();

  for (int kb = 0; kb < Dc / 32; ++kb) {
    const int k0 = kb * 32;
    if (kb + 8 < Dc / 32) {  // near-cache, non-speculative prefetch of streams
      __builtin_prefetch(A  + (long)(m0 + (lane & 15)) * Dc + k0 + 256, 0, 3);
      __builtin_prefetch(Wt + (long)(n0 + (lane & 15)) * Dc + k0 + 256, 0, 3);
    }
    v16bf a[4];
#pragma unroll
    for (int i = 0; i < 4; ++i) a[i] = load_a_frag(A, Dc, m0 + i * 16, k0);
#pragma unroll
    for (int j = 0; j < 4; ++j) {
      v16bf bj = load_b_frag(Wt, Dc, n0 + j * 16, k0);
#pragma unroll
      for (int i = 0; i < 4; ++i) acc[i][j] = WMMA_BF16(a[i], bj, acc[i][j]);
    }
  }

  // C/D layout: element r in lane l -> row = r + (l>=16 ? 8 : 0), col = l%16
  const int half = lane >> 4, cn = lane & 15;
#pragma unroll
  for (int i = 0; i < 4; ++i) {
#pragma unroll
    for (int r = 0; r < 8; ++r) {
      const int grow = m0 + i * 16 + r + half * 8;
      const int b = grow / Sc, s = grow % Sc;
#pragma unroll
      for (int j = 0; j < 4; ++j) {
        const int col = n0 + j * 16 + cn;
        const float v = (acc[i][j][r] + bias[col]) * scale;
        if (mode == 0) {
          ((unsigned short*)out)[((long)b * obStride + oRowOff + s) * ldOut + col] =
              f32_to_bf16(v);
        } else if (mode == 1) {
          ((unsigned short*)out)[((long)b * Dc + col) * ldOut + oRowOff + s] =
              f32_to_bf16(v);
        } else {
          ((float*)out)[((long)b * obStride + s) * ldOut + col] = v;
        }
      }
    }
  }
}

// ---------------------------------------------------------------------------
// Flash attention: 4 waves x 16 queries; 64-key blocks over KP=4096 keys.
// Scores via WMMA (Q pre-scaled by 1/sqrt(HD)), online softmax with in-half
// shuffle reductions, P transposed through per-wave LDS, P*V via WMMA.
// ---------------------------------------------------------------------------
__global__ void __launch_bounds__(128) attn_flash(
    const unsigned short* __restrict__ Q, const unsigned short* __restrict__ Kb,
    const unsigned short* __restrict__ Vt, unsigned short* __restrict__ O) {
  __shared__ unsigned short Ptile[4][16 * 64];

  const int lane = threadIdx.x & 31;
  const int wid  = threadIdx.x >> 5;
  const int half = lane >> 4, ln = lane & 15;
  const int bh = blockIdx.y, b = bh / Hc, h = bh % Hc;
  const int qbase = blockIdx.x * 64 + wid * 16;

  v16bf qf[4];
#pragma unroll
  for (int kc = 0; kc < 4; ++kc)
    qf[kc] = load_a_frag(Q, Dc, b * Sc + qbase, h * HDc + kc * 32);

  v8f accO[8];
#pragma unroll
  for (int n = 0; n < 8; ++n) accO[n] = zero_v8f();
  float mrun[8], lrun[8];
#pragma unroll
  for (int r = 0; r < 8; ++r) { mrun[r] = -1e30f; lrun[r] = 0.0f; }

  for (int kb = 0; kb < KPc / 64; ++kb) {
    // --- scores: 16x64 tile = 4 accumulators, QK^T over HD=128 (4 k-chunks)
    v8f sc[4];
#pragma unroll
    for (int j = 0; j < 4; ++j) {
      sc[j] = zero_v8f();
      const int krow0 = b * KPc + kb * 64 + j * 16;
#pragma unroll
      for (int kc = 0; kc < 4; ++kc) {
        v16bf kf = load_b_frag(Kb, Dc, krow0, h * HDc + kc * 32);
        sc[j] = WMMA_BF16(qf[kc], kf, sc[j]);
      }
    }
    // --- row max (rows 0-7 live in lanes 0-15, rows 8-15 in lanes 16-31)
    float cur[8];
#pragma unroll
    for (int r = 0; r < 8; ++r) {
      float m = fmaxf(fmaxf(sc[0][r], sc[1][r]), fmaxf(sc[2][r], sc[3][r]));
#pragma unroll
      for (int msk = 1; msk < 16; msk <<= 1) m = fmaxf(m, __shfl_xor(m, msk, 32));
      cur[r] = m;
    }
    float sf[8], rs[8];
#pragma unroll
    for (int r = 0; r < 8; ++r) {
      const float nm = fmaxf(mrun[r], cur[r]);
      sf[r] = __expf(mrun[r] - nm);
      mrun[r] = nm;
      rs[r] = 0.0f;
    }
    // --- p = exp(s - m); stash bf16 P tile in per-wave LDS for transpose
#pragma unroll
    for (int j = 0; j < 4; ++j)
#pragma unroll
      for (int r = 0; r < 8; ++r) {
        const float p = __expf(sc[j][r] - mrun[r]);
        rs[r] += p;
        Ptile[wid][(r + half * 8) * 64 + j * 16 + ln] = f32_to_bf16(p);
      }
#pragma unroll
    for (int r = 0; r < 8; ++r) {
      float s = rs[r];
#pragma unroll
      for (int msk = 1; msk < 16; msk <<= 1) s += __shfl_xor(s, msk, 32);
      lrun[r] = lrun[r] * sf[r] + s;
    }
#pragma unroll
    for (int n = 0; n < 8; ++n)
#pragma unroll
      for (int r = 0; r < 8; ++r) accO[n][r] *= sf[r];

    // --- O += P @ V  (P as A-fragments from LDS; V from transposed Vt)
    v16bf pf[2];
#pragma unroll
    for (int kc = 0; kc < 2; ++kc)
      pf[kc] = load_a_frag(&Ptile[wid][0], 64, 0, kc * 32);
    const unsigned short* vbase = Vt + (long)b * Dc * KPc;
#pragma unroll
    for (int n = 0; n < 8; ++n)
#pragma unroll
      for (int kc = 0; kc < 2; ++kc) {
        v16bf vf = load_b_frag(vbase, KPc, h * HDc + n * 16, kb * 64 + kc * 32);
        accO[n] = WMMA_BF16(pf[kc], vf, accO[n]);
      }
  }

  float inv[8];
#pragma unroll
  for (int r = 0; r < 8; ++r) inv[r] = 1.0f / lrun[r];
#pragma unroll
  for (int n = 0; n < 8; ++n)
#pragma unroll
    for (int r = 0; r < 8; ++r) {
      const int row = b * Sc + qbase + r + half * 8;
      const int col = h * HDc + n * 16 + ln;
      O[(long)row * Dc + col] = f32_to_bf16(accO[n][r] * inv[r]);
    }
}

// ---------------------------------------------------------------------------
extern "C" void kernel_launch(void* const* d_in, const int* in_sizes, int n_in,
                              void* d_out, int out_size, void* d_ws, size_t ws_size,
                              hipStream_t stream) {
  (void)in_sizes; (void)n_in; (void)out_size; (void)ws_size;
  const float* x  = (const float*)d_in[0];
  const float* ck = (const float*)d_in[1];
  const float* cv = (const float*)d_in[2];
  const float* wq = (const float*)d_in[3];
  const float* bq = (const float*)d_in[4];
  const float* wk = (const float*)d_in[5];
  const float* bk = (const float*)d_in[6];
  const float* wv = (const float*)d_in[7];
  const float* bv = (const float*)d_in[8];
  const float* wo = (const float*)d_in[9];
  const float* bo = (const float*)d_in[10];
  float* out = (float*)d_out;

  char* ws = (char*)d_ws;
  size_t off = 0;
  auto take = [&](size_t elems) {
    void* p = ws + off;
    off += (elems * 2 + 255) & ~(size_t)255;
    return (unsigned short*)p;
  };
  unsigned short* XBF = take((size_t)Mc * Dc);        //  16 MB
  unsigned short* QBF = take((size_t)Mc * Dc);        //  16 MB
  unsigned short* KBF = take((size_t)Bc * KPc * Dc);  //  32 MB
  unsigned short* VT  = take((size_t)Bc * Dc * KPc);  //  32 MB (transposed V)
  unsigned short* OBF = take((size_t)Mc * Dc);        //  16 MB
  unsigned short* WTQ = take((size_t)Dc * Dc);        //   8 MB each
  unsigned short* WTK = take((size_t)Dc * Dc);
  unsigned short* WTV = take((size_t)Dc * Dc);
  unsigned short* WTO = take((size_t)Dc * Dc);

  const int thr = 256;
  const long nx = (long)Mc * Dc;
  f32_to_bf16_kernel<<<dim3((unsigned)((nx + thr - 1) / thr)), thr, 0, stream>>>(x, XBF, nx);
  const long nw = (long)Dc * Dc;
  transpose_w_kernel<<<dim3((unsigned)((nw + thr - 1) / thr)), thr, 0, stream>>>(wq, WTQ);
  transpose_w_kernel<<<dim3((unsigned)((nw + thr - 1) / thr)), thr, 0, stream>>>(wk, WTK);
  transpose_w_kernel<<<dim3((unsigned)((nw + thr - 1) / thr)), thr, 0, stream>>>(wv, WTV);
  transpose_w_kernel<<<dim3((unsigned)((nw + thr - 1) / thr)), thr, 0, stream>>>(wo, WTO);
  const long nc = (long)Bc * Pc * Dc;
  cache_k_kernel<<<dim3((unsigned)((nc + thr - 1) / thr)), thr, 0, stream>>>(ck, KBF);
  cache_v_kernel<<<dim3((unsigned)((nc + thr - 1) / thr)), thr, 0, stream>>>(cv, VT);

  const dim3 ggrid(Dc / 256, Mc / 128);   // 128x256 block tiles
  const float qscale = 0.08838834764831845f;  // 1/sqrt(HD), folded into Q
  gemm_bf16<<<ggrid, 256, 0, stream>>>(XBF, WTQ, bq, QBF, qscale, 0, Sc, 0, Dc);
  gemm_bf16<<<ggrid, 256, 0, stream>>>(XBF, WTK, bk, KBF, 1.0f, 0, KPc, Pc, Dc);
  gemm_bf16<<<ggrid, 256, 0, stream>>>(XBF, WTV, bv, VT, 1.0f, 1, 0, Pc, KPc);

  attn_flash<<<dim3(Sc / 64, Bc * Hc), 128, 0, stream>>>(QBF, KBF, VT, OBF);

  gemm_bf16<<<ggrid, 256, 0, stream>>>(OBF, WTO, bo, out, 1.0f, 2, Sc, 0, Dc);
}
